// SuperGlue_3882650436631
// MI455X (gfx1250) — matrix operand order repaired and checked by reading, not compiled, not run on hardware
//
#include <hip/hip_runtime.h>

typedef _Float16 half8  __attribute__((ext_vector_type(8)));
typedef _Float16 half16 __attribute__((ext_vector_type(16)));
typedef float    float8 __attribute__((ext_vector_type(8)));

constexpr int   BATCH = 2;
constexpr int   N     = 2048;      // keypoints per image
constexpr int   D     = 256;       // descriptor dim
constexpr int   NP    = 2049;      // N + dustbin
constexpr float NORM       = -8.31776617f;  // -log(N+N) = -log(4096)
constexpr float LOG_CORNER = -0.69314718f;  // log(2048) - log(4096)
constexpr float SCALE      = 0.0625f;       // 1/sqrt(256)
constexpr float MATCH_THRESH = 0.2f;

// ---------------------------------------------------------------- prep
__global__ void cvt_f16_k(const float* __restrict__ src, _Float16* __restrict__ dst, int n) {
    int i = blockIdx.x * blockDim.x + threadIdx.x;
    if (i < n) dst[i] = (_Float16)src[i];
}

__global__ void wtrans_f16_k(const float* __restrict__ W, _Float16* __restrict__ WT) {
    int i = blockIdx.x * blockDim.x + threadIdx.x;   // i = d*256 + j
    int d = i >> 8, j = i & 255;
    WT[(size_t)j * D + d] = (_Float16)W[i];
}

// ------------------------------------------------- projection: mdesc = desc@W + b
// A: (BATCH*N, D) f16 row-major.  WT: (D, D) f16, WT[j][d] = W[d][j].
// One wave per 16x16 output tile; 8 waves per block.
__global__ __launch_bounds__(256) void proj_wmma_k(const _Float16* __restrict__ A,
                                                   const _Float16* __restrict__ WT,
                                                   const float* __restrict__ bias,
                                                   _Float16* __restrict__ C) {
    const int wave = blockIdx.x * 8 + (threadIdx.x >> 5);
    const int lane = threadIdx.x & 31;
    const int hw = lane >> 4, l16 = lane & 15;
    const int tm = wave >> 4, tn = wave & 15;       // 256 row-tiles x 16 col-tiles
    const int m0 = tm * 16, n0 = tn * 16;

    float8 acc = {};
    const _Float16* ap = A  + (size_t)(m0 + l16) * D + 8  * hw;
    const _Float16* bp = WT + (size_t)(n0 + l16) * D + 8  * hw;
    for (int k = 0; k < D; k += 32) {
        union { half16 v; half8 h[2]; } a;
        a.h[0] = *(const half8*)(ap + k);        // K = k + 8hw .. +7
        a.h[1] = *(const half8*)(ap + k + 16);   // K = k + 16 + 8hw .. +7
        half16 bf = __builtin_shufflevector(*(const half8*)(bp + k), *(const half8*)(bp + k + 16), 0,1,2,3,4,5,6,7,8,9,10,11,12,13,14,15);    // K = k + 8hw .. +7, k + 16 + 8hw .. +7, N = l16
        acc = __builtin_amdgcn_wmma_f32_16x16x32_f16(false, a.v, false, bf,
                                                     (short)0, acc, false, false);
    }
    const int col = n0 + l16;
    const float bb = bias[col];
    for (int r = 0; r < 8; ++r) {
        const int row = m0 + r + 8 * hw;
        C[(size_t)row * D + col] = (_Float16)(acc[r] + bb);
    }
}

// --------------------------------------- scores: Z = mdesc0 . mdesc1^T / 16
// Bt (=mdesc1) is already the row-major transpose of the B operand.
// Writes both Z and ZT so both Sinkhorn directions sweep coalesced rows.
__global__ __launch_bounds__(256) void scores_wmma_k(const _Float16* __restrict__ M0,
                                                     const _Float16* __restrict__ M1,
                                                     float* __restrict__ Z,
                                                     float* __restrict__ ZT) {
    const int wave = blockIdx.x * 8 + (threadIdx.x >> 5);
    const int lane = threadIdx.x & 31;
    const int hw = lane >> 4, l16 = lane & 15;
    const int b = wave >> 14;                 // 16384 = 128*128 tiles per batch
    const int t = wave & 16383;
    const int ti = t >> 7, tj = t & 127;
    const int i0 = ti * 16, j0 = tj * 16;

    const _Float16* A  = M0 + (size_t)b * N * D;
    const _Float16* Bt = M1 + (size_t)b * N * D;
    float8 acc = {};
    const _Float16* ap = A  + (size_t)(i0 + l16) * D + 8  * hw;
    const _Float16* bp = Bt + (size_t)(j0 + l16) * D + 8  * hw;
    for (int k = 0; k < D; k += 32) {
        union { half16 v; half8 h[2]; } a;
        a.h[0] = *(const half8*)(ap + k);
        a.h[1] = *(const half8*)(ap + k + 16);
        half16 bf = __builtin_shufflevector(*(const half8*)(bp + k), *(const half8*)(bp + k + 16), 0,1,2,3,4,5,6,7,8,9,10,11,12,13,14,15);
        acc = __builtin_amdgcn_wmma_f32_16x16x32_f16(false, a.v, false, bf,
                                                     (short)0, acc, false, false);
    }
    float* Zb  = Z  + (size_t)b * NP * NP;
    float* ZTb = ZT + (size_t)b * NP * NP;
    const int j = j0 + l16;
    for (int r = 0; r < 8; ++r) {
        const int i = i0 + r + 8 * hw;
        const float s = acc[r] * SCALE;
        Zb [(size_t)i * NP + j] = s;
        ZTb[(size_t)j * NP + i] = s;
    }
}

// --------------------------- dustbin row/col fill + u,v zero init
__global__ void init_edges_k(float* __restrict__ Z, float* __restrict__ ZT,
                             float* __restrict__ u, float* __restrict__ v,
                             const float* __restrict__ bin) {
    int i = blockIdx.x * blockDim.x + threadIdx.x;
    if (i >= BATCH * NP) return;
    int b = i / NP, r = i % NP;
    float alpha = bin[0];
    float* Zb  = Z  + (size_t)b * NP * NP;
    float* ZTb = ZT + (size_t)b * NP * NP;
    Zb [(size_t)r * NP + (NP - 1)] = alpha;
    Zb [(size_t)(NP - 1) * NP + r] = alpha;
    ZTb[(size_t)r * NP + (NP - 1)] = alpha;
    ZTb[(size_t)(NP - 1) * NP + r] = alpha;
    u[i] = 0.0f; v[i] = 0.0f;
}

// --------------------------- one Sinkhorn half-step: uout = log_mu - LSE_j(Z[r,j]+vin[j])
__global__ __launch_bounds__(256) void lse_row_k(const float* __restrict__ Zm,
                                                 const float* __restrict__ vin,
                                                 float* __restrict__ uout) {
    __shared__ float red[256];
    const int b = blockIdx.x / NP;
    const int r = blockIdx.x % NP;
    const float* row = Zm + ((size_t)b * NP + r) * NP;
    const float* vv  = vin + (size_t)b * NP;
    const int tid = threadIdx.x;

    float m = -3.4e38f;
    for (int j = tid; j < NP; j += 256) m = fmaxf(m, row[j] + vv[j]);
    red[tid] = m; __syncthreads();
    for (int s = 128; s > 0; s >>= 1) {
        if (tid < s) red[tid] = fmaxf(red[tid], red[tid + s]);
        __syncthreads();
    }
    m = red[0]; __syncthreads();

    float sum = 0.0f;
    for (int j = tid; j < NP; j += 256) sum += __expf(row[j] + vv[j] - m);
    red[tid] = sum; __syncthreads();
    for (int s = 128; s > 0; s >>= 1) {
        if (tid < s) red[tid] += red[tid + s];
        __syncthreads();
    }
    if (tid == 0) {
        float target = (r < N) ? NORM : LOG_CORNER;
        uout[(size_t)b * NP + r] = target - (m + __logf(red[0]));
    }
}

// --------------------------- row argmax over inner part (first-occurrence ties)
__global__ __launch_bounds__(256) void rowargmax_k(const float* __restrict__ Zm,
                                                   const float* __restrict__ other,
                                                   const float* __restrict__ own,
                                                   float* __restrict__ outmax,
                                                   int* __restrict__ outidx) {
    __shared__ float rv[256];
    __shared__ int   ri[256];
    const int b = blockIdx.x / N;
    const int r = blockIdx.x % N;
    const float* row = Zm + ((size_t)b * NP + r) * NP;
    const float* ov  = other + (size_t)b * NP;
    const int tid = threadIdx.x;

    float m = -3.4e38f; int mi = 0;
    for (int j = tid; j < N; j += 256) {          // exclude dustbin column
        float vv = row[j] + ov[j];
        if (vv > m) { m = vv; mi = j; }           // j ascending -> first max kept
    }
    rv[tid] = m; ri[tid] = mi; __syncthreads();
    for (int s = 128; s > 0; s >>= 1) {
        if (tid < s) {
            float v2 = rv[tid + s]; int i2 = ri[tid + s];
            if (v2 > rv[tid] || (v2 == rv[tid] && i2 < ri[tid])) { rv[tid] = v2; ri[tid] = i2; }
        }
        __syncthreads();
    }
    if (tid == 0) {
        outmax[(size_t)b * N + r] = rv[0] + own[(size_t)b * NP + r] - NORM;
        outidx[(size_t)b * N + r] = ri[0];
    }
}

// --------------------------- mutual matching
__global__ void match_k(const float* __restrict__ max0, const int* __restrict__ idx0,
                        const int* __restrict__ idx1,
                        int* __restrict__ mo0, int* __restrict__ mo1,
                        float* __restrict__ so0, float* __restrict__ so1) {
    int i = blockIdx.x * blockDim.x + threadIdx.x;
    if (i >= BATCH * N) return;
    int b = i / N, r = i % N;
    const int* i0 = idx0 + (size_t)b * N;
    const int* i1 = idx1 + (size_t)b * N;
    { // role 0: row r of image-0
        int j = i0[r];
        bool mutual = (i1[j] == r);
        float ms = mutual ? __expf(max0[i]) : 0.0f;
        bool valid = mutual && (ms > MATCH_THRESH);
        mo0[i] = valid ? j : -1;
        so0[i] = ms;
    }
    { // role 1: column r of image-1 (mutual1 implies mutual0[idx1[r]])
        int k = i1[r];
        bool mutual = (i0[k] == r);
        float msk = __expf(max0[(size_t)b * N + k]);
        float ms = mutual ? msk : 0.0f;
        bool valid = mutual && (msk > MATCH_THRESH);
        mo1[i] = valid ? k : -1;
        so1[i] = ms;
    }
}

// ----------------------------------------------------------------------
extern "C" void kernel_launch(void* const* d_in, const int* in_sizes, int n_in,
                              void* d_out, int out_size, void* d_ws, size_t ws_size,
                              hipStream_t stream) {
    (void)in_sizes; (void)n_in; (void)out_size; (void)ws_size;
    const float* desc0 = (const float*)d_in[0];
    const float* desc1 = (const float*)d_in[1];
    const float* W     = (const float*)d_in[2];
    const float* bias  = (const float*)d_in[3];
    const float* binp  = (const float*)d_in[4];

    char* ws = (char*)d_ws;
    auto alloc = [&](size_t bytes) {
        char* p = ws;
        ws += (bytes + 255) & ~(size_t)255;
        return (void*)p;
    };
    _Float16* d0h = (_Float16*)alloc((size_t)BATCH * N * D * 2);
    _Float16* d1h = (_Float16*)alloc((size_t)BATCH * N * D * 2);
    _Float16* WT  = (_Float16*)alloc((size_t)D * D * 2);
    _Float16* md0 = (_Float16*)alloc((size_t)BATCH * N * D * 2);
    _Float16* md1 = (_Float16*)alloc((size_t)BATCH * N * D * 2);
    float* Z    = (float*)alloc((size_t)BATCH * NP * NP * 4);
    float* ZT   = (float*)alloc((size_t)BATCH * NP * NP * 4);
    float* u    = (float*)alloc((size_t)BATCH * NP * 4);
    float* v    = (float*)alloc((size_t)BATCH * NP * 4);
    float* max0 = (float*)alloc((size_t)BATCH * N * 4);
    float* max1 = (float*)alloc((size_t)BATCH * N * 4);
    int*   idx0 = (int*)alloc((size_t)BATCH * N * 4);
    int*   idx1 = (int*)alloc((size_t)BATCH * N * 4);

    const int nElem = BATCH * N * D;                      // 1,048,576
    cvt_f16_k<<<nElem / 256, 256, 0, stream>>>(desc0, d0h, nElem);
    cvt_f16_k<<<nElem / 256, 256, 0, stream>>>(desc1, d1h, nElem);
    wtrans_f16_k<<<(D * D) / 256, 256, 0, stream>>>(W, WT);

    proj_wmma_k<<<512, 256, 0, stream>>>(d0h, WT, bias, md0);   // 4096 tiles
    proj_wmma_k<<<512, 256, 0, stream>>>(d1h, WT, bias, md1);
    scores_wmma_k<<<4096, 256, 0, stream>>>(md0, md1, Z, ZT);   // 32768 tiles

    init_edges_k<<<(BATCH * NP + 255) / 256, 256, 0, stream>>>(Z, ZT, u, v, binp);

    for (int it = 0; it < 100; ++it) {
        lse_row_k<<<BATCH * NP, 256, 0, stream>>>(Z,  v, u);
        lse_row_k<<<BATCH * NP, 256, 0, stream>>>(ZT, u, v);
    }

    rowargmax_k<<<BATCH * N, 256, 0, stream>>>(Z,  v, u, max0, idx0);
    rowargmax_k<<<BATCH * N, 256, 0, stream>>>(ZT, u, v, max1, idx1);

    int*   mo0 = (int*)d_out;
    int*   mo1 = mo0 + BATCH * N;
    float* so0 = (float*)d_out + 2 * BATCH * N;
    float* so1 = (float*)d_out + 3 * BATCH * N;
    match_k<<<(BATCH * N + 255) / 256, 256, 0, stream>>>(max0, idx0, idx1, mo0, mo1, so0, so1);
}
